// FlexibleInertialEncoder_9964324127421
// MI455X (gfx1250) — matrix operand order repaired and checked by reading, not compiled, not run on hardware
//
#include <hip/hip_runtime.h>
#include <hip/hip_bf16.h>

// ---------------------------------------------------------------------------
// FlexibleInertialEncoder for MI455X (gfx1250, wave32, WMMA)
//
// Pipeline: conv1(6->64,VALU f32) -> conv2(64->128, bf16 WMMA) ->
//           conv3(128->256, bf16 WMMA, fused pooling) -> FC1 -> FC2
// BN folded into weights/bias by a prep kernel. Compute-bound (~36 GFLOP,
// ~5MB traffic) => use v_wmma_f32_16x16x32_bf16 (K=32/instr vs K=4 for f32).
// conv2/conv3 keep all A (weight) fragments in registers across the N-tile
// loop; only B activations stream from LDS (2x ds_load_b128 per wmma).
// ---------------------------------------------------------------------------

typedef __attribute__((ext_vector_type(4)))  unsigned int v4u;
typedef __attribute__((ext_vector_type(8)))  float        v8f;
typedef __attribute__((ext_vector_type(16))) __bf16       v16bf;

union Frag { v4u q[2]; v16bf bf; };

__device__ __forceinline__ unsigned short f2bf(float f) {
  unsigned int u = __float_as_uint(f);
  u += 0x7FFFu + ((u >> 16) & 1u);           // round-to-nearest-even
  return (unsigned short)(u >> 16);
}

// A-matrix 16x32 bf16 fragment (CDNA5 ISA "16-bit A-Matrix 16x32"):
// lane L: M = L%16; halves i<8 -> K = (L>=16?8:0)+i ; i>=8 -> K = 16+(L>=16?8:0)+(i-8)
// => two contiguous 16B chunks at K0 = kbase + hi*8 and K0+16.
__device__ __forceinline__ v16bf load_a_frag(const unsigned short* rowk, int hi) {
  Frag f;
  f.q[0] = *(const v4u*)(rowk + hi * 8);
  f.q[1] = *(const v4u*)(rowk + hi * 8 + 16);
  return f.bf;
}

// B-matrix 32x16 bf16 fragment (same K order as A: i<8 -> K = hi*8+i,
// i>=8 -> K = 16+hi*8+(i-8)). Value source indexed [N][K] (i.e.
// transposed activations / row-major weight rows), two 16B chunks per lane.
__device__ __forceinline__ v16bf load_b_frag(const unsigned short* nrow_k, int hi) {
  Frag f;
  const unsigned short* p = nrow_k + hi * 8;
  f.q[0] = *(const v4u*)(p);
  f.q[1] = *(const v4u*)(p + 16);
  return f.bf;
}

__device__ __forceinline__ v8f wmma_bf16(v16bf a, v16bf b, v8f c) {
  return __builtin_amdgcn_wmma_f32_16x16x32_bf16(false, a, false, b, (short)0, c,
                                                 false, false);
}

// ---------------------------------------------------------------------------
// Workspace layout (bytes, all 256-aligned)
// ---------------------------------------------------------------------------
#define O_W1F   ((size_t)0)          // [64][18] f32 folded conv1 weights (4608 B)
#define O_B1    ((size_t)4608)       // [64]  f32 folded bias
#define O_B2    ((size_t)4864)       // [128] f32
#define O_B3    ((size_t)5376)       // [256] f32
#define O_W2A   ((size_t)6400)       // [3][128][64]  bf16 (49152 B)
#define O_W3A   ((size_t)55552)      // [3][256][128] bf16 (196608 B)
#define O_LW1B  ((size_t)252160)     // [512][3328] bf16 (3407872 B)
#define O_LW2B  ((size_t)3660032)    // [256][512]  bf16 (262144 B)
#define O_MS    ((size_t)3922176)    // [1024][3328] bf16 pooled features (6815744 B)
#define O_H2FC  ((size_t)10737920)   // [1024][512] bf16 FC1 output (1048576 B)
// total ~11.8 MB

// ---------------------------------------------------------------------------
// Prep: fold BN into conv weights/biases, pack per-tap matrices, cvt to bf16
// ---------------------------------------------------------------------------
#define PREP_TOTAL (1152 + 64 + 128 + 256 + 24576 + 98304 + 1703936 + 131072)

__global__ void prep_kernel(
    const float* __restrict__ w1, const float* __restrict__ b1,
    const float* __restrict__ w2, const float* __restrict__ b2,
    const float* __restrict__ w3, const float* __restrict__ b3,
    const float* __restrict__ g1, const float* __restrict__ be1,
    const float* __restrict__ m1, const float* __restrict__ v1,
    const float* __restrict__ g2, const float* __restrict__ be2,
    const float* __restrict__ m2, const float* __restrict__ v2,
    const float* __restrict__ g3, const float* __restrict__ be3,
    const float* __restrict__ m3, const float* __restrict__ v3,
    const float* __restrict__ lw1, const float* __restrict__ lw2,
    float* __restrict__ w1f, float* __restrict__ b1f,
    float* __restrict__ b2f, float* __restrict__ b3f,
    unsigned short* __restrict__ w2a, unsigned short* __restrict__ w3a,
    unsigned short* __restrict__ lw1b, unsigned short* __restrict__ lw2b) {
  const int stride = gridDim.x * blockDim.x;
  for (int idx = blockIdx.x * blockDim.x + threadIdx.x; idx < PREP_TOTAL;
       idx += stride) {
    int i = idx;
    if (i < 1152) {                       // w1 fold (f32, layout [co][ci*3+dt])
      int co = i / 18;
      w1f[i] = w1[i] * (g1[co] * rsqrtf(v1[co] + 1e-5f));
      continue;
    }
    i -= 1152;
    if (i < 64) {
      float s = g1[i] * rsqrtf(v1[i] + 1e-5f);
      b1f[i] = (b1[i] - m1[i]) * s + be1[i];
      continue;
    }
    i -= 64;
    if (i < 128) {
      float s = g2[i] * rsqrtf(v2[i] + 1e-5f);
      b2f[i] = (b2[i] - m2[i]) * s + be2[i];
      continue;
    }
    i -= 128;
    if (i < 256) {
      float s = g3[i] * rsqrtf(v3[i] + 1e-5f);
      b3f[i] = (b3[i] - m3[i]) * s + be3[i];
      continue;
    }
    i -= 256;
    if (i < 24576) {                      // w2 -> 3 per-tap [128][64] bf16
      int dt = i / 8192, r = i % 8192, co = r >> 6, ci = r & 63;
      float s = g2[co] * rsqrtf(v2[co] + 1e-5f);
      w2a[i] = f2bf(w2[co * 192 + ci * 3 + dt] * s);
      continue;
    }
    i -= 24576;
    if (i < 98304) {                      // w3 -> 3 per-tap [256][128] bf16
      int dt = i / 32768, r = i % 32768, co = r >> 7, ci = r & 127;
      float s = g3[co] * rsqrtf(v3[co] + 1e-5f);
      w3a[i] = f2bf(w3[co * 384 + ci * 3 + dt] * s);
      continue;
    }
    i -= 98304;
    if (i < 1703936) { lw1b[i] = f2bf(lw1[i]); continue; }
    i -= 1703936;
    lw2b[i] = f2bf(lw2[i]);
  }
}

// ---------------------------------------------------------------------------
// Conv stack + fused pooling: one block (8 waves) per sequence.
// LDS: transposed bf16 activations h1T[t+1][cin] / h2T[t+1][cin] with zeroed
// halo rows; padded strides for conflict-free ds_load_b128.
// ---------------------------------------------------------------------------
#define S1 72    // 64 + 8 pad  (bf16 elems per row)
#define S2 136   // 128 + 8 pad

__global__ __launch_bounds__(256) void conv_kernel(
    const float* __restrict__ x,               // [1024][128][6]
    const float* __restrict__ w1f,             // [64][18] folded
    const float* __restrict__ b1f, const float* __restrict__ b2f,
    const float* __restrict__ b3f,
    const unsigned short* __restrict__ w2a,    // [3][128][64]  bf16
    const unsigned short* __restrict__ w3a,    // [3][256][128] bf16
    unsigned short* __restrict__ ms) {         // [1024][3328]  bf16
  __shared__ __align__(16) unsigned short h2T[130 * S2];  // 35360 B
  __shared__ __align__(16) unsigned short h1T[130 * S1];  // 18720 B
  __shared__ float lsum[256 * 8];                         //  8192 B
  __shared__ float b2ls[128];
  __shared__ float b3ls[256];

  const int tid  = threadIdx.x;
  const int n    = blockIdx.x;         // sequence index
  const int lane = tid & 31;
  const int wid  = tid >> 5;
  const int hi   = (lane >> 4) & 1;
  const int l16  = lane & 15;

  // ---- phase 0: init LDS ----
  for (int i = tid; i < 2048; i += 256) lsum[i] = 0.0f;
  if (tid < 128) b2ls[tid] = b2f[tid];
  b3ls[tid] = b3f[tid];
  for (int i = tid; i < S1; i += 256) { h1T[i] = 0; h1T[129 * S1 + i] = 0; }
  for (int i = tid; i < S2; i += 256) { h2T[i] = 0; h2T[129 * S2 + i] = 0; }
  __syncthreads();

  // ---- phase 1: conv1 (6->64) in f32 VALU, write bf16 h1T[t+1][c] ----
  {
    const float* xn = x + (size_t)n * 768;   // [128][6]
    const int t = tid & 127;
    const int cblk = tid >> 7;               // 0 or 1 (32 channels each)
    float xv[3][6];
#pragma unroll
    for (int dt = 0; dt < 3; ++dt) {
      int tt = t - 1 + dt;
      bool ok = (tt >= 0) && (tt < 128);
#pragma unroll
      for (int ci = 0; ci < 6; ++ci) xv[dt][ci] = ok ? xn[tt * 6 + ci] : 0.0f;
    }
    for (int cc = 0; cc < 32; ++cc) {
      int c = cblk * 32 + cc;
      const float* wr = w1f + c * 18;
      float acc = b1f[c];
#pragma unroll
      for (int ci = 0; ci < 6; ++ci)
#pragma unroll
        for (int dt = 0; dt < 3; ++dt)
          acc = fmaf(wr[ci * 3 + dt], xv[dt][ci], acc);
      acc = fmaxf(acc, 0.1f * acc);          // leaky relu
      h1T[(t + 1) * S1 + c] = f2bf(acc);
    }
  }
  __syncthreads();

  // ---- phase 2: conv2 (64->128) = sum over 3 taps of [128x64]x[64x128] ----
  // wave 'wid' owns M-tile mt=wid; 6 A fragments preloaded in registers,
  // inner loop over 8 N-tiles streams B from LDS. 48 WMMA per wave.
  {
    const int mt = wid;
    v16bf Afr[6];
#pragma unroll
    for (int dt = 0; dt < 3; ++dt) {
      const unsigned short* Ad = w2a + dt * (128 * 64) + (mt * 16 + l16) * 64;
#pragma unroll
      for (int kk = 0; kk < 2; ++kk)
        Afr[dt * 2 + kk] = load_a_frag(Ad + kk * 32, hi);
    }
    for (int nt = 0; nt < 8; ++nt) {
      v8f acc = {};
#pragma unroll
      for (int dt = 0; dt < 3; ++dt) {
        const unsigned short* Bd = &h1T[(nt * 16 + l16 + dt) * S1];
#pragma unroll
        for (int kk = 0; kk < 2; ++kk)
          acc = wmma_bf16(Afr[dt * 2 + kk], load_b_frag(Bd + kk * 32, hi), acc);
      }
      int t = nt * 16 + l16;
      unsigned int pk[4];
#pragma unroll
      for (int r = 0; r < 8; r += 2) {
        int c0 = mt * 16 + hi * 8 + r;
        float v0 = acc[r] + b2ls[c0];         v0 = fmaxf(v0, 0.1f * v0);
        float v1 = acc[r + 1] + b2ls[c0 + 1]; v1 = fmaxf(v1, 0.1f * v1);
        pk[r >> 1] = (unsigned int)f2bf(v0) | ((unsigned int)f2bf(v1) << 16);
      }
      v4u st; st[0] = pk[0]; st[1] = pk[1]; st[2] = pk[2]; st[3] = pk[3];
      *(v4u*)&h2T[(t + 1) * S2 + mt * 16 + hi * 8] = st;   // 16B aligned
    }
  }
  __syncthreads();

  // ---- phase 3: conv3 (128->256) fused with pooling ----
  // Wave owns 2 M-tiles (mt = 2*wid + half). For each, the 12 A fragments
  // (96 VGPRs) are loaded once and reused across all 8 N-tiles. Each 16-wide
  // N tile is exactly one local pooling bin: cross-lane reduce the 16-lane
  // half and accumulate bin sums in LDS; h3 is never materialized.
  for (int half = 0; half < 2; ++half) {
    const int mt = wid * 2 + half;
    v16bf Afr[12];
#pragma unroll
    for (int dt = 0; dt < 3; ++dt) {
      const unsigned short* Ad = w3a + dt * (256 * 128) + (mt * 16 + l16) * 128;
#pragma unroll
      for (int kk = 0; kk < 4; ++kk)
        Afr[dt * 4 + kk] = load_a_frag(Ad + kk * 32, hi);
    }
    for (int nt = 0; nt < 8; ++nt) {
      v8f acc = {};
#pragma unroll
      for (int dt = 0; dt < 3; ++dt) {
        const unsigned short* Bd = &h2T[(nt * 16 + l16 + dt) * S2];
#pragma unroll
        for (int kk = 0; kk < 4; ++kk)
          acc = wmma_bf16(Afr[dt * 4 + kk], load_b_frag(Bd + kk * 32, hi), acc);
      }
#pragma unroll
      for (int r = 0; r < 8; ++r) {
        int c = mt * 16 + hi * 8 + r;
        float v = acc[r] + b3ls[c];
        v = fmaxf(v, 0.1f * v);
        v += __shfl_xor(v, 1, 32);           // sum over the 16 T-columns
        v += __shfl_xor(v, 2, 32);
        v += __shfl_xor(v, 4, 32);
        v += __shfl_xor(v, 8, 32);
        if (l16 == 0) atomicAdd(&lsum[c * 8 + nt], v);
      }
    }
  }
  __syncthreads();

  // ---- phase 4: pooled feature vector (local/mid/glob from bin sums) ----
  {
    int c = tid;  // 256 channels, 256 threads
    float ls[8];
    float g = 0.0f;
#pragma unroll
    for (int b = 0; b < 8; ++b) { ls[b] = lsum[c * 8 + b]; g += ls[b]; }
    unsigned short* mp = ms + (size_t)n * 3328;
#pragma unroll
    for (int b = 0; b < 8; ++b) mp[c * 8 + b] = f2bf(ls[b] * 0.0625f);
#pragma unroll
    for (int j = 0; j < 4; ++j)
      mp[2048 + c * 4 + j] = f2bf((ls[2 * j] + ls[2 * j + 1]) * 0.03125f);
    mp[3072 + c] = f2bf(g * 0.0078125f);
  }
}

// ---------------------------------------------------------------------------
// FC1: h2 = relu(ms[1024,3328] @ lw1^T + lb1) -> bf16 [1024][512]
// One wave per 16x16 tile; 2048 tiles; K loop of 104 bf16 WMMAs.
// ---------------------------------------------------------------------------
__global__ __launch_bounds__(256) void fc1_kernel(
    const unsigned short* __restrict__ ms,     // [1024][3328]
    const unsigned short* __restrict__ lw1b,   // [512][3328]
    const float* __restrict__ lb1,
    unsigned short* __restrict__ h2fc) {       // [1024][512]
  const int tid = threadIdx.x, lane = tid & 31, wid = tid >> 5;
  const int hi = (lane >> 4) & 1, l16 = lane & 15;
  const int tile = blockIdx.x * 8 + wid;       // < 2048
  const int mt = tile >> 5, nt = tile & 31;
  const unsigned short* Ap = ms + (size_t)(mt * 16 + l16) * 3328;
  const unsigned short* Bp = lw1b + (size_t)(nt * 16 + l16) * 3328;
  v8f acc = {};
  for (int kk = 0; kk < 104; ++kk) {
    v16bf a = load_a_frag(Ap + kk * 32, hi);
    v16bf b = load_b_frag(Bp + kk * 32, hi);
    acc = wmma_bf16(a, b, acc);
  }
  float bias = lb1[nt * 16 + l16];
#pragma unroll
  for (int r = 0; r < 8; ++r) {
    int m = mt * 16 + hi * 8 + r;
    float v = fmaxf(acc[r] + bias, 0.0f);
    h2fc[(size_t)m * 512 + nt * 16 + l16] = f2bf(v);
  }
}

// ---------------------------------------------------------------------------
// FC2: out = h2 @ lw2^T + lb2 -> f32 [1024][256]
// ---------------------------------------------------------------------------
__global__ __launch_bounds__(256) void fc2_kernel(
    const unsigned short* __restrict__ h2fc,   // [1024][512]
    const unsigned short* __restrict__ lw2b,   // [256][512]
    const float* __restrict__ lb2,
    float* __restrict__ out) {                 // [1024][256]
  const int tid = threadIdx.x, lane = tid & 31, wid = tid >> 5;
  const int hi = (lane >> 4) & 1, l16 = lane & 15;
  const int tile = blockIdx.x * 8 + wid;       // < 1024
  const int mt = tile >> 4, nt = tile & 15;
  const unsigned short* Ap = h2fc + (size_t)(mt * 16 + l16) * 512;
  const unsigned short* Bp = lw2b + (size_t)(nt * 16 + l16) * 512;
  v8f acc = {};
#pragma unroll 4
  for (int kk = 0; kk < 16; ++kk) {
    v16bf a = load_a_frag(Ap + kk * 32, hi);
    v16bf b = load_b_frag(Bp + kk * 32, hi);
    acc = wmma_bf16(a, b, acc);
  }
  float bias = lb2[nt * 16 + l16];
#pragma unroll
  for (int r = 0; r < 8; ++r) {
    int m = mt * 16 + hi * 8 + r;
    out[(size_t)m * 256 + nt * 16 + l16] = acc[r] + bias;
  }
}

// ---------------------------------------------------------------------------
extern "C" void kernel_launch(void* const* d_in, const int* in_sizes, int n_in,
                              void* d_out, int out_size, void* d_ws,
                              size_t ws_size, hipStream_t stream) {
  const float* x   = (const float*)d_in[0];
  const float* w1  = (const float*)d_in[1];
  const float* b1  = (const float*)d_in[2];
  const float* w2  = (const float*)d_in[3];
  const float* b2  = (const float*)d_in[4];
  const float* w3  = (const float*)d_in[5];
  const float* b3  = (const float*)d_in[6];
  const float* g1  = (const float*)d_in[7];
  const float* be1 = (const float*)d_in[8];
  const float* m1  = (const float*)d_in[9];
  const float* v1  = (const float*)d_in[10];
  const float* g2  = (const float*)d_in[11];
  const float* be2 = (const float*)d_in[12];
  const float* m2  = (const float*)d_in[13];
  const float* v2  = (const float*)d_in[14];
  const float* g3  = (const float*)d_in[15];
  const float* be3 = (const float*)d_in[16];
  const float* m3  = (const float*)d_in[17];
  const float* v3  = (const float*)d_in[18];
  const float* lw1 = (const float*)d_in[19];
  const float* lb1 = (const float*)d_in[20];
  const float* lw2 = (const float*)d_in[21];
  const float* lb2 = (const float*)d_in[22];

  char* ws = (char*)d_ws;
  float* w1f = (float*)(ws + O_W1F);
  float* b1f = (float*)(ws + O_B1);
  float* b2f = (float*)(ws + O_B2);
  float* b3f = (float*)(ws + O_B3);
  unsigned short* w2a  = (unsigned short*)(ws + O_W2A);
  unsigned short* w3a  = (unsigned short*)(ws + O_W3A);
  unsigned short* lw1b = (unsigned short*)(ws + O_LW1B);
  unsigned short* lw2b = (unsigned short*)(ws + O_LW2B);
  unsigned short* msb  = (unsigned short*)(ws + O_MS);
  unsigned short* h2fc = (unsigned short*)(ws + O_H2FC);

  prep_kernel<<<2048, 256, 0, stream>>>(w1, b1, w2, b2, w3, b3, g1, be1, m1,
                                        v1, g2, be2, m2, v2, g3, be3, m3, v3,
                                        lw1, lw2, w1f, b1f, b2f, b3f, w2a, w3a,
                                        lw1b, lw2b);
  conv_kernel<<<1024, 256, 0, stream>>>(x, w1f, b1f, b2f, b3f, w2a, w3a, msb);
  fc1_kernel<<<256, 256, 0, stream>>>(msb, lw1b, lb1, h2fc);
  fc2_kernel<<<128, 256, 0, stream>>>(h2fc, lw2b, lb2, (float*)d_out);
}
